// MultiHeadAttnBlock_20074677141954
// MI455X (gfx1250) — matrix operand order repaired and checked by reading, not compiled, not run on hardware
//
#include <hip/hip_runtime.h>
#include <hip/hip_bf16.h>

// ---------------------------------------------------------------------------
// MultiHeadAttnBlock for MI455X (gfx1250, wave32, WMMA f16 16x16x32 -> f32)
//  B=8, C=512, H=W=32 (n=1024 tokens), heads=8, head_dim=64, groups=32
// Pipeline: groupnorm->f16, wcvt, qkv gemm (async double-buffered, K-slab 64),
//           flash attention (async K staging), out-proj+residual
// ---------------------------------------------------------------------------

typedef __attribute__((ext_vector_type(16))) _Float16 v16h;
typedef __attribute__((ext_vector_type(8)))  float    v8f;

#define C_CH   512
#define NTOK   1024
#define BATCH  8
#define HEADS  8
#define ADIM   64
#define CPG    16        // channels per group (512/32)
#define NGRP   32

#define MT     128       // GEMM M tile
#define NT     64        // GEMM N tile
#define KS     64        // GEMM staged K slab (2 x 32 WMMA steps)
#define XLD    72        // padded LDS ld (halves) for A tiles
#define WLD    72        // padded LDS ld (halves) for B tiles

#define KVT    128       // attention KV tile (keys)
#define KLD    68        // padded ld for K tile in LDS
#define PLD    136       // padded ld for per-wave P staging

static __device__ __forceinline__ v8f zero8() {
  v8f z;
#pragma unroll
  for (int i = 0; i < 8; ++i) z[i] = 0.0f;
  return z;
}

// CDNA5 async memory->LDS copy (8 bytes/lane), GVS form:
//   mem = SGPR_base + VGPR_u32_offset ; LDS byte addr from VDST VGPR.
// Per-thread offsets stay loop-invariant; only the uniform SGPR base moves.
static __device__ __forceinline__ void async_b64_s(unsigned lds_off, unsigned goff,
                                                   const _Float16* sbase) {
  asm volatile("global_load_async_to_lds_b64 %0, %1, %2"
               :
               : "v"(lds_off), "v"(goff), "s"(sbase)
               : "memory");
}
static __device__ __forceinline__ void wait_async0() {
  asm volatile("s_wait_asynccnt 0x0" ::: "memory");
}
// low 32 bits of a generic pointer to __shared__ == LDS byte offset
static __device__ __forceinline__ unsigned lds_off_of(const _Float16* p) {
  return (unsigned)(unsigned long long)p;
}

// A-matrix fragment (16x32 f16), row-major source with leading dim ld (halves).
// Lane l holds row (l&15); element s holds K = 16*(l>>4) + s.
static __device__ __forceinline__ v16h load_a_frag(const _Float16* __restrict__ src, int ld) {
  const int lane = threadIdx.x & 31;
  const _Float16* p = src + (lane & 15) * ld + (lane >> 4) * 16;
  v16h a;
#pragma unroll
  for (int j = 0; j < 16; ++j) a[j] = p[j];
  return a;
}

// B-matrix fragment (32x16 f16) from B-transposed (row-major [N][K]) source:
// lane l holds column (l&15), K = (l>>4)*16 .. +15  -> 16 contiguous halves.
static __device__ __forceinline__ v16h load_b_frag(const _Float16* __restrict__ srcT, int ld) {
  const int lane = threadIdx.x & 31;
  const _Float16* p = srcT + (size_t)(lane & 15) * ld + (lane >> 4) * 16;
  v16h b;
#pragma unroll
  for (int j = 0; j < 16; ++j) b[j] = p[j];
  return b;
}

static __device__ __forceinline__ v8f wmma16(v16h a, v16h b, v8f c) {
  return __builtin_amdgcn_wmma_f32_16x16x32_f16(false, a, false, b, (short)0, c,
                                                false, false);
}

// ---------------------------------------------------------------------------
// Kernel 1: GroupNorm over (C/32, H, W); output f16 token-major [b*n, C]
// ---------------------------------------------------------------------------
__global__ void gn_kernel(const float* __restrict__ x, const float* __restrict__ gam,
                          const float* __restrict__ bet, _Float16* __restrict__ hn) {
  const int bg = blockIdx.x;                 // 0..255 = (b, group)
  const int b = bg >> 5, grp = bg & 31;
  const float* xb = x + (size_t)b * C_CH * NTOK + (size_t)grp * CPG * NTOK;

  float s = 0.f, ss = 0.f;
  for (int idx = threadIdx.x; idx < CPG * NTOK; idx += blockDim.x) {
    const int i = idx & 15, n = idx >> 4;
    const float v = xb[i * NTOK + n];
    s += v; ss += v * v;
  }
#pragma unroll
  for (int o = 16; o >= 1; o >>= 1) {
    s  += __shfl_xor(s, o, 32);
    ss += __shfl_xor(ss, o, 32);
  }
  __shared__ float rs[8], rss[8], stat[2];
  const int wid = threadIdx.x >> 5;
  if ((threadIdx.x & 31) == 0) { rs[wid] = s; rss[wid] = ss; }
  __syncthreads();
  if (threadIdx.x == 0) {
    float S = 0.f, SS = 0.f;
#pragma unroll
    for (int i = 0; i < 8; ++i) { S += rs[i]; SS += rss[i]; }
    const float inv = 1.0f / (float)(CPG * NTOK);
    const float mu = S * inv;
    const float var = SS * inv - mu * mu;
    stat[0] = mu;
    stat[1] = rsqrtf(var + 1e-6f);
  }
  __syncthreads();
  const float mu = stat[0], rstd = stat[1];
  for (int idx = threadIdx.x; idx < CPG * NTOK; idx += blockDim.x) {
    const int i = idx & 15, n = idx >> 4;
    const int c = grp * CPG + i;
    const float v = (xb[i * NTOK + n] - mu) * rstd * gam[c] + bet[c];
    hn[((size_t)b * NTOK + n) * C_CH + c] = (_Float16)v;
  }
}

// ---------------------------------------------------------------------------
// Kernel 2: weights f32 -> f16 (wq,wk,wv,wo), kept row-major [o][c]
// ---------------------------------------------------------------------------
__global__ void wcvt_kernel(const float* __restrict__ wq, const float* __restrict__ wk,
                            const float* __restrict__ wv, const float* __restrict__ wo,
                            _Float16* __restrict__ wf) {
  const float* srcs[4] = {wq, wk, wv, wo};
  const int stride = gridDim.x * blockDim.x;
  const int t0 = blockIdx.x * blockDim.x + threadIdx.x;
#pragma unroll
  for (int m = 0; m < 4; ++m) {
    const float* s = srcs[m];
    _Float16* d = wf + (size_t)m * C_CH * C_CH;
    for (int i = t0; i < C_CH * C_CH; i += stride) d[i] = (_Float16)s[i];
  }
}

// ---------------------------------------------------------------------------
// Kernel 3: QKV projections. M=8192 tokens, N=512, K=512. Double-buffered
// async staging of 64-wide K slabs; 8 WMMAs per barrier.
//  mat 0 -> Q token-major [bh][n][a] (scale folded), mat 1 -> K token-major,
//  mat 2 -> V channel-major [bh][a][n].
// ---------------------------------------------------------------------------
__global__ void qkv_gemm_kernel(const _Float16* __restrict__ hn,
                                const _Float16* __restrict__ wf,
                                const float* __restrict__ bq, const float* __restrict__ bk,
                                const float* __restrict__ bv,
                                _Float16* __restrict__ Qt, _Float16* __restrict__ Kt,
                                _Float16* __restrict__ Vt) {
  __shared__ _Float16 Xs[2 * MT * XLD];
  __shared__ _Float16 Ws[2 * NT * WLD];

  const int mat = blockIdx.z;
  const int m0 = blockIdx.x * MT;
  const int n0 = blockIdx.y * NT;
  const _Float16* W = wf + (size_t)mat * C_CH * C_CH;      // [o][c] == B^T
  const float* bias = (mat == 0) ? bq : (mat == 1 ? bk : bv);

  const int tid = threadIdx.x, wid = tid >> 5, lane = tid & 31;
  const int lane15 = lane & 15, hlf = lane >> 4;
  const int wm = wid & 3, wn = wid >> 2;                   // 4x2 waves of 32x32

  // loop-invariant per-thread staging offsets (16 b64-chunks per 64-half row)
  unsigned gX[8], lX[8], gW[4], lW[4];
#pragma unroll
  for (int u = 0; u < 8; ++u) {
    const int i = tid + u * 256;
    const int r = i >> 4, c = i & 15;
    gX[u] = (unsigned)((r * C_CH + c * 4) * 2);
    lX[u] = lds_off_of(Xs + r * XLD + c * 4);
  }
#pragma unroll
  for (int u = 0; u < 4; ++u) {
    const int i = tid + u * 256;
    const int r = i >> 4, c = i & 15;
    gW[u] = (unsigned)((r * C_CH + c * 4) * 2);
    lW[u] = lds_off_of(Ws + r * WLD + c * 4);
  }
  const _Float16* baseX = hn + (size_t)m0 * C_CH;
  const _Float16* baseW = W + (size_t)n0 * C_CH;

  auto stage = [&](int buf, int k0) {
    const unsigned ox = buf ? (unsigned)(MT * XLD * 2) : 0u;
    const unsigned ow = buf ? (unsigned)(NT * WLD * 2) : 0u;
    const _Float16* sx = baseX + k0;
    const _Float16* sw = baseW + k0;
#pragma unroll
    for (int u = 0; u < 8; ++u) async_b64_s(lX[u] + ox, gX[u], sx);
#pragma unroll
    for (int u = 0; u < 4; ++u) async_b64_s(lW[u] + ow, gW[u], sw);
  };

  v8f acc[2][2];
#pragma unroll
  for (int i = 0; i < 2; ++i)
#pragma unroll
    for (int j = 0; j < 2; ++j) acc[i][j] = zero8();

  stage(0, 0);
  wait_async0();
  __syncthreads();

  for (int ks = 0; ks < C_CH / KS; ++ks) {
    const int cur = ks & 1;
    if (ks + 1 < C_CH / KS) stage(cur ^ 1, (ks + 1) * KS);
    const _Float16* Xc = Xs + cur * (MT * XLD);
    const _Float16* Wc = Ws + cur * (NT * WLD);

#pragma unroll
    for (int kk = 0; kk < 2; ++kk) {
      v16h afr[2], bfr[2];
      afr[0] = load_a_frag(Xc + (wm * 32 + 0)  * XLD + kk * 32, XLD);
      afr[1] = load_a_frag(Xc + (wm * 32 + 16) * XLD + kk * 32, XLD);
      bfr[0] = load_b_frag(Wc + (wn * 32 + 0)  * WLD + kk * 32, WLD);
      bfr[1] = load_b_frag(Wc + (wn * 32 + 16) * WLD + kk * 32, WLD);
#pragma unroll
      for (int i = 0; i < 2; ++i)
#pragma unroll
        for (int j = 0; j < 2; ++j) acc[i][j] = wmma16(afr[i], bfr[j], acc[i][j]);
    }
    wait_async0();
    __syncthreads();
  }

  // epilogue: bias (+ attention scale folded into Q), scatter to head layouts
  const float qscale = 0.00552427172f;    // (64*512)^-0.5
#pragma unroll
  for (int ti = 0; ti < 2; ++ti) {
#pragma unroll
    for (int tj = 0; tj < 2; ++tj) {
      const int col = n0 + wn * 32 + tj * 16 + lane15;   // output channel
      const float bsv = bias[col];
      const int hh = col >> 6, aa = col & 63;
#pragma unroll
      for (int j = 0; j < 8; ++j) {
        const int row = m0 + wm * 32 + ti * 16 + j + 8 * hlf;   // token
        const int bi = row >> 10, nt = row & 1023;
        float v = acc[ti][tj][j] + bsv;
        if (mat == 0) v *= qscale;
        if (mat == 2) {
          Vt[(((size_t)(bi * HEADS + hh)) * ADIM + aa) * NTOK + nt] = (_Float16)v;
        } else {
          _Float16* dst = (mat == 0) ? Qt : Kt;
          dst[(((size_t)(bi * HEADS + hh)) * NTOK + nt) * ADIM + aa] = (_Float16)v;
        }
      }
    }
  }
}

// ---------------------------------------------------------------------------
// Kernel 4: flash attention. Block = (qtile, b*h); 8 waves x 16 query rows.
// ---------------------------------------------------------------------------
__global__ void attn_kernel(const _Float16* __restrict__ Qt, const _Float16* __restrict__ Kt,
                            const _Float16* __restrict__ Vt, _Float16* __restrict__ X2) {
  __shared__ _Float16 Ks[KVT * KLD];          // K tile, token-major [m][a]
  __shared__ _Float16 Ps[8 * 16 * PLD];       // per-wave P staging [16][128]

  const int qt = blockIdx.x;                  // 0..7
  const int bh = blockIdx.y;                  // 0..63
  const int tid = threadIdx.x, wid = tid >> 5, lane = tid & 31;
  const int lane15 = lane & 15, hlf = lane >> 4;

  const _Float16* Qg = Qt + (size_t)bh * NTOK * ADIM;
  const _Float16* Kg = Kt + (size_t)bh * NTOK * ADIM;
  const _Float16* Vg = Vt + (size_t)bh * ADIM * NTOK;

  // loop-invariant K-tile staging offsets (8 b64-chunks per key row: only a 0..31 of the 64)
  unsigned gK[4], lK[4];
#pragma unroll
  for (int u = 0; u < 4; ++u) {
    const int i = tid + u * 256;
    const int r = i >> 3, c = i & 7;
    gK[u] = (unsigned)((r * ADIM + c * 4) * 2);
    lK[u] = lds_off_of(Ks + r * KLD + c * 4);
  }

  const int qrow0 = qt * 128 + wid * 16;
  v16h aq[2];
  aq[0] = load_a_frag(Qg + (size_t)qrow0 * ADIM + 0,  ADIM);
  aq[1] = load_a_frag(Qg + (size_t)qrow0 * ADIM + 32, ADIM);

  float mrow[8], lrow[8];
  v8f oacc[4];
#pragma unroll
  for (int j = 0; j < 8; ++j) { mrow[j] = -3.0e38f; lrow[j] = 0.f; }
#pragma unroll
  for (int c = 0; c < 4; ++c) oacc[c] = zero8();

  _Float16* Pw = Ps + (size_t)wid * 16 * PLD;

  for (int m0 = 0; m0 < NTOK; m0 += KVT) {
    // async-stage K tile [128 keys][a 0..31 only]
    const _Float16* sk = Kg + (size_t)m0 * ADIM;
#pragma unroll
    for (int u = 0; u < 4; ++u) async_b64_s(lK[u], gK[u], sk);
    wait_async0();
    __syncthreads();

    // scores: Q[16x64] x K^T -> 16x128 (8 column tiles, k split 2x32)
    v8f sacc[8];
#pragma unroll
    for (int t = 0; t < 8; ++t) sacc[t] = zero8();
#pragma unroll
    for (int t = 0; t < 8; ++t) {
#pragma unroll
      for (int kk = 0; kk < 2; ++kk) {
        const v16h bkf = load_b_frag(Ks + (t * 16) * KLD + kk * 32, KLD);
        sacc[t] = wmma16(aq[kk], bkf, sacc[t]);
      }
    }

    // online softmax: VGPR j <-> row (j + 8*hlf); cols spread over 16 lanes
    float mnew[8], scl[8], ssum[8];
#pragma unroll
    for (int j = 0; j < 8; ++j) {
      float mx = sacc[0][j];
#pragma unroll
      for (int t = 1; t < 8; ++t) mx = fmaxf(mx, sacc[t][j]);
      mx = fmaxf(mx, __shfl_xor(mx, 1, 32));
      mx = fmaxf(mx, __shfl_xor(mx, 2, 32));
      mx = fmaxf(mx, __shfl_xor(mx, 4, 32));
      mx = fmaxf(mx, __shfl_xor(mx, 8, 32));
      mnew[j] = fmaxf(mrow[j], mx);
      scl[j] = __expf(mrow[j] - mnew[j]);
      mrow[j] = mnew[j];
      ssum[j] = 0.f;
    }
#pragma unroll
    for (int t = 0; t < 8; ++t) {
#pragma unroll
      for (int j = 0; j < 8; ++j) {
        const float p = __expf(sacc[t][j] - mnew[j]);
        ssum[j] += p;
        Pw[(j + 8 * hlf) * PLD + t * 16 + lane15] = (_Float16)p;
      }
    }
#pragma unroll
    for (int j = 0; j < 8; ++j) {
      float s = ssum[j];
      s += __shfl_xor(s, 1, 32);
      s += __shfl_xor(s, 2, 32);
      s += __shfl_xor(s, 4, 32);
      s += __shfl_xor(s, 8, 32);
      lrow[j] = lrow[j] * scl[j] + s;
#pragma unroll
      for (int c = 0; c < 4; ++c) oacc[c][j] *= scl[j];
    }
    asm volatile("s_wait_dscnt 0x0" ::: "memory");  // P staged for this wave

    // O += P[16x128] x V[128x64]; V^T ([a][n]) streamed from L2 as B fragments
#pragma unroll
    for (int c = 0; c < 4; ++c) {
#pragma unroll
      for (int kk = 0; kk < 4; ++kk) {
        const v16h ap = load_a_frag(Pw + kk * 32, PLD);
        const v16h bvf = load_b_frag(Vg + (size_t)(c * 16) * NTOK + m0 + kk * 32, NTOK);
        oacc[c] = wmma16(ap, bvf, oacc[c]);
      }
    }
    __syncthreads();
  }

  // normalize and write token-major [b*n][C] at channel base h*64
  const int b = bh >> 3, h = bh & 7;
#pragma unroll
  for (int c = 0; c < 4; ++c) {
#pragma unroll
    for (int j = 0; j < 8; ++j) {
      const int row = qrow0 + j + 8 * hlf;
      const float v = oacc[c][j] / lrow[j];
      X2[((size_t)b * NTOK + row) * C_CH + h * ADIM + c * 16 + lane15] = (_Float16)v;
    }
  }
}

// ---------------------------------------------------------------------------
// Kernel 5: output projection + bias + residual; f32 out [b][c][n]
// ---------------------------------------------------------------------------
__global__ void proj_kernel(const _Float16* __restrict__ X2, const _Float16* __restrict__ woT,
                            const float* __restrict__ bo, const float* __restrict__ res,
                            float* __restrict__ out) {
  __shared__ _Float16 Xs[2 * MT * XLD];
  __shared__ _Float16 Ws[2 * NT * WLD];

  const int m0 = blockIdx.x * MT;
  const int n0 = blockIdx.y * NT;
  const int tid = threadIdx.x, wid = tid >> 5, lane = tid & 31;
  const int lane15 = lane & 15, hlf = lane >> 4;
  const int wm = wid & 3, wn = wid >> 2;

  unsigned gX[8], lX[8], gW[4], lW[4];
#pragma unroll
  for (int u = 0; u < 8; ++u) {
    const int i = tid + u * 256;
    const int r = i >> 4, c = i & 15;
    gX[u] = (unsigned)((r * C_CH + c * 4) * 2);
    lX[u] = lds_off_of(Xs + r * XLD + c * 4);
  }
#pragma unroll
  for (int u = 0; u < 4; ++u) {
    const int i = tid + u * 256;
    const int r = i >> 4, c = i & 15;
    gW[u] = (unsigned)((r * C_CH + c * 4) * 2);
    lW[u] = lds_off_of(Ws + r * WLD + c * 4);
  }
  const _Float16* baseX = X2 + (size_t)m0 * C_CH;
  const _Float16* baseW = woT + (size_t)n0 * C_CH;

  auto stage = [&](int buf, int k0) {
    const unsigned ox = buf ? (unsigned)(MT * XLD * 2) : 0u;
    const unsigned ow = buf ? (unsigned)(NT * WLD * 2) : 0u;
    const _Float16* sx = baseX + k0;
    const _Float16* sw = baseW + k0;
#pragma unroll
    for (int u = 0; u < 8; ++u) async_b64_s(lX[u] + ox, gX[u], sx);
#pragma unroll
    for (int u = 0; u < 4; ++u) async_b64_s(lW[u] + ow, gW[u], sw);
  };

  v8f acc[2][2];
#pragma unroll
  for (int i = 0; i < 2; ++i)
#pragma unroll
    for (int j = 0; j < 2; ++j) acc[i][j] = zero8();

  stage(0, 0);
  wait_async0();
  __syncthreads();

  for (int ks = 0; ks < C_CH / KS; ++ks) {
    const int cur = ks & 1;
    if (ks + 1 < C_CH / KS) stage(cur ^ 1, (ks + 1) * KS);
    const _Float16* Xc = Xs + cur * (MT * XLD);
    const _Float16* Wc = Ws + cur * (NT * WLD);

#pragma unroll
    for (int kk = 0; kk < 2; ++kk) {
      v16h afr[2], bfr[2];
      afr[0] = load_a_frag(Xc + (wm * 32 + 0)  * XLD + kk * 32, XLD);
      afr[1] = load_a_frag(Xc + (wm * 32 + 16) * XLD + kk * 32, XLD);
      bfr[0] = load_b_frag(Wc + (wn * 32 + 0)  * WLD + kk * 32, WLD);
      bfr[1] = load_b_frag(Wc + (wn * 32 + 16) * WLD + kk * 32, WLD);
#pragma unroll
      for (int i = 0; i < 2; ++i)
#pragma unroll
        for (int j = 0; j < 2; ++j) acc[i][j] = wmma16(afr[i], bfr[j], acc[i][j]);
    }
    wait_async0();
    __syncthreads();
  }

#pragma unroll
  for (int ti = 0; ti < 2; ++ti) {
#pragma unroll
    for (int tj = 0; tj < 2; ++tj) {
      const int col = n0 + wn * 32 + tj * 16 + lane15;
      const float bsv = bo[col];
#pragma unroll
      for (int j = 0; j < 8; ++j) {
        const int row = m0 + wm * 32 + ti * 16 + j + 8 * hlf;
        const int bi = row >> 10, nt = row & 1023;
        const size_t idx = ((size_t)bi * C_CH + col) * NTOK + nt;
        out[idx] = acc[ti][tj][j] + bsv + res[idx];
      }
    }
  }
}

// ---------------------------------------------------------------------------
extern "C" void kernel_launch(void* const* d_in, const int* in_sizes, int n_in,
                              void* d_out, int out_size, void* d_ws, size_t ws_size,
                              hipStream_t stream) {
  const float* kv  = (const float*)d_in[0];
  const float* gg  = (const float*)d_in[1];
  const float* gb  = (const float*)d_in[2];
  const float* wq  = (const float*)d_in[3];
  const float* bq  = (const float*)d_in[4];
  const float* wk  = (const float*)d_in[5];
  const float* bk  = (const float*)d_in[6];
  const float* wv  = (const float*)d_in[7];
  const float* bv  = (const float*)d_in[8];
  const float* wo  = (const float*)d_in[9];
  const float* bo  = (const float*)d_in[10];
  float* out = (float*)d_out;

  char* ws = (char*)d_ws;
  // workspace layout (bytes), ~44 MB total
  _Float16* hn = (_Float16*)(ws + 0);                    //  8 MB  [b*n][C] f16
  _Float16* wf = (_Float16*)(ws + 8388608);              //  2 MB  4x[o][c] f16
  _Float16* Qt = (_Float16*)(ws + 10485760);             //  8 MB  [bh][n][a]
  _Float16* Kt = (_Float16*)(ws + 18874368);             //  8 MB  [bh][n][a]
  _Float16* Vt = (_Float16*)(ws + 27262976);             //  8 MB  [bh][a][n]
  _Float16* X2 = (_Float16*)(ws + 35651584);             //  8 MB  [b*n][C]

  gn_kernel<<<dim3(BATCH * NGRP), dim3(256), 0, stream>>>(kv, gg, gb, hn);
  wcvt_kernel<<<dim3(256), dim3(256), 0, stream>>>(wq, wk, wv, wo, wf);
  qkv_gemm_kernel<<<dim3(64, 8, 3), dim3(256), 0, stream>>>(hn, wf, bq, bk, bv,
                                                            Qt, Kt, Vt);
  attn_kernel<<<dim3(8, 64), dim3(256), 0, stream>>>(Qt, Kt, Vt, X2);
  proj_kernel<<<dim3(64, 8), dim3(256), 0, stream>>>(
      X2, wf + (size_t)3 * C_CH * C_CH, bo, kv, out);
}
